// TemporalMeshGCN_55851754717430
// MI455X (gfx1250) — matrix operand order repaired and checked by reading, not compiled, not run on hardware
//
#include <hip/hip_runtime.h>
#include <hip/hip_bf16.h>

typedef __attribute__((ext_vector_type(16))) _Float16 v16h;
typedef __attribute__((ext_vector_type(8)))  float    v8f;

#define NB 4
#define TT 25
#define NN 10000
#define FF 3
#define CC 16
#define BN (NB*NN)          // 40000 rows
#define HSZ (BN*CC)         // 640000 floats per hidden buffer
#define NTILE (BN/16)       // 2500 WMMA row-tiles

// fast transcendentals: v_rcp_f32 / v_exp_f32, no IEEE div sequences
__device__ __forceinline__ float fast_sigmoid(float x) {
  return __builtin_amdgcn_rcpf(1.f + __expf(-x));
}
__device__ __forceinline__ float fast_tanh(float x) {
  // 1 - 2/(e^{2x}+1); saturates correctly at +/-1 for large |x|
  return 1.f - 2.f * __builtin_amdgcn_rcpf(__expf(2.f * x) + 1.f);
}

// ---------------- utility ----------------
__global__ void k_fill(float* __restrict__ p, float v, int n) {
  int i = blockIdx.x * blockDim.x + threadIdx.x;
  int stride = gridDim.x * blockDim.x;
  for (; i < n; i += stride) p[i] = v;
}

// ---------------- graph normalization ----------------
__global__ void k_degscatter(const long long* __restrict__ ei,
                             const float* __restrict__ ew,
                             float* __restrict__ deg, int E) {
  int e = blockIdx.x * blockDim.x + threadIdx.x;
  if (e >= E) return;
  atomicAdd(&deg[(int)ei[E + e]], ew[e]);   // dst row of edge_index
}

__global__ void k_rsqrt(float* __restrict__ deg) {
  int i = blockIdx.x * blockDim.x + threadIdx.x;
  if (i >= NN) return;
  float d = deg[i];
  deg[i] = d > 0.f ? rsqrtf(d) : 0.f;
}

__global__ void k_norm(const long long* __restrict__ ei,
                       const float* __restrict__ ew,
                       const float* __restrict__ dinv,
                       float* __restrict__ nrm, int E) {
  int e = blockIdx.x * blockDim.x + threadIdx.x;
  if (e >= E + NN) return;
  if (e < E) {
    nrm[e] = dinv[(int)ei[e]] * ew[e] * dinv[(int)ei[E + e]];
  } else {
    float d = dinv[e - E];      // self loop, weight 1
    nrm[e] = d * d;
  }
}

// ---------------- GRU weights: (32x16 f32, k-major) -> f16 column-major per gate
// LT[g*512 + n*32 + k]: a lane's B-fragment (16 consecutive k for column n)
// is one contiguous 32-byte load.
__global__ void k_cvtw(const float* __restrict__ Lz, const float* __restrict__ Lr,
                       const float* __restrict__ Lh, _Float16* __restrict__ LT) {
  int idx = blockIdx.x * blockDim.x + threadIdx.x;
  if (idx >= 3 * 512) return;
  int g = idx >> 9;
  int r = idx & 511;
  int n = r >> 5;     // 0..15 output column
  int k = r & 31;     // 0..31 K row
  const float* L = (g == 0) ? Lz : (g == 1) ? Lr : Lh;
  LT[g * 512 + n * 32 + k] = (_Float16)L[k * 16 + n];
}

// ---------------- per-step: s = x_t @ W for all three gates ----------------
__global__ void k_xw(const float* __restrict__ x,
                     const float* __restrict__ Wz, const float* __restrict__ Wr,
                     const float* __restrict__ Wh,
                     float* __restrict__ sz, float* __restrict__ sr,
                     float* __restrict__ sh, int t) {
  int idx = blockIdx.x * blockDim.x + threadIdx.x;   // (b, n) flat
  if (idx >= BN) return;
  int b = idx / NN, nn = idx - b * NN;
  const float* xp = x + (((size_t)b * TT + t) * NN + nn) * FF;
  float x0 = xp[0], x1 = xp[1], x2 = xp[2];
  float* pz = sz + (size_t)idx * CC;
  float* pr = sr + (size_t)idx * CC;
  float* ph = sh + (size_t)idx * CC;
#pragma unroll
  for (int c = 0; c < CC; ++c) {
    pz[c] = x0 * Wz[c] + x1 * Wz[16 + c] + x2 * Wz[32 + c];
    pr[c] = x0 * Wr[c] + x1 * Wr[16 + c] + x2 * Wr[32 + c];
    ph[c] = x0 * Wh[c] + x1 * Wh[16 + c] + x2 * Wh[32 + c];
  }
}

// ---------------- per-step: edge gather/scale/scatter for all gates ----------
__global__ void k_scatter(const long long* __restrict__ ei,
                          const float* __restrict__ nrm,
                          const float* __restrict__ sz, const float* __restrict__ sr,
                          const float* __restrict__ sh,
                          float* __restrict__ cz, float* __restrict__ cr,
                          float* __restrict__ ch, int E) {
  int idx = blockIdx.x * blockDim.x + threadIdx.x;
  int tot = (E + NN) * CC;
  if (idx >= tot) return;
  int c = idx & 15;
  int e = idx >> 4;
  int s, d;
  if (e < E) { s = (int)ei[e]; d = (int)ei[E + e]; }
  else       { s = d = e - E; }
  float w = nrm[e];
#pragma unroll
  for (int b = 0; b < NB; ++b) {
    int so = b * (NN * CC) + s * CC + c;
    int dd = b * (NN * CC) + d * CC + c;
    atomicAdd(&cz[dd], sz[so] * w);
    atomicAdd(&cr[dd], sr[so] * w);
    atomicAdd(&ch[dd], sh[so] * w);
  }
}

// ---------------- fused GRU step: one wave32 per 16-row tile, 3 WMMAs --------
__global__ void k_gru(const float* __restrict__ cz, const float* __restrict__ cr,
                      const float* __restrict__ ch,
                      const float* __restrict__ bz, const float* __restrict__ br,
                      const float* __restrict__ bh,
                      const _Float16* __restrict__ LT,
                      const float* __restrict__ lbz, const float* __restrict__ lbr,
                      const float* __restrict__ lbh,
                      const float* __restrict__ H, float* __restrict__ Hn) {
  __shared__ float ldsR[8 * 256];                // one 16x16 R tile per wave
  const int lane = threadIdx.x & 31;
  const int wave = threadIdx.x >> 5;
  const int tile = blockIdx.x * 8 + wave;
  const bool valid = tile < NTILE;
  const int rowBase = (valid ? tile : 0) * 16;
  const int m  = lane & 15;
  const int hi = lane >> 4;
  const int rowA = rowBase + m;

  // B fragments: column n=m, K = hi*16 .. hi*16+15 contiguous
  const v16h bzf = *(const v16h*)(LT +    0 + m * 32 + hi * 16);
  const v16h brf = *(const v16h*)(LT +  512 + m * 32 + hi * 16);
  const v16h bhf = *(const v16h*)(LT + 1024 + m * 32 + hi * 16);

  // A fragments for Z and R gates: cols 0..15 = conv+bias, 16..31 = H
  v16h az, ar;
#pragma unroll
  for (int j = 0; j < 8; ++j) {
    const int k0 = 2 * j + hi * 16;
    if (k0 < 16) {                                // conv+bias columns
      az[2*j]   = (_Float16)(cz[rowA*16 + k0    ] + bz[k0]);
      az[2*j+1] = (_Float16)(cz[rowA*16 + k0 + 1] + bz[k0+1]);
      ar[2*j]   = (_Float16)(cr[rowA*16 + k0    ] + br[k0]);
      ar[2*j+1] = (_Float16)(cr[rowA*16 + k0 + 1] + br[k0+1]);
    } else {                                      // H columns (k0 >= 16)
      const int c = k0 - 16;
      _Float16 h0 = (_Float16)H[rowA*16 + c];
      _Float16 h1 = (_Float16)H[rowA*16 + c + 1];
      az[2*j] = h0; az[2*j+1] = h1;
      ar[2*j] = h0; ar[2*j+1] = h1;
    }
  }

  v8f ciz, cir;
  const float lzv = lbz[m], lrv = lbr[m];
#pragma unroll
  for (int j = 0; j < 8; ++j) { ciz[j] = lzv; cir[j] = lrv; }

  v8f Zd = __builtin_amdgcn_wmma_f32_16x16x32_f16(false, az, false, bzf,
                                                  (short)0, ciz, false, false);
  v8f Rd = __builtin_amdgcn_wmma_f32_16x16x32_f16(false, ar, false, brf,
                                                  (short)0, cir, false, false);

  float zs[8];
#pragma unroll
  for (int j = 0; j < 8; ++j) {
    zs[j] = fast_sigmoid(Zd[j]);
    float rv = fast_sigmoid(Rd[j]);
    ldsR[wave * 256 + (j + hi * 8) * 16 + m] = rv;   // D layout -> row-major
  }
  __syncthreads();

  // A fragment for candidate gate: cols 0..15 = conv_h+bias, 16..31 = H*R
  v16h ah;
#pragma unroll
  for (int j = 0; j < 8; ++j) {
    const int k0 = 2 * j + hi * 16;
    if (k0 < 16) {
      ah[2*j]   = (_Float16)(ch[rowA*16 + k0    ] + bh[k0]);
      ah[2*j+1] = (_Float16)(ch[rowA*16 + k0 + 1] + bh[k0+1]);
    } else {
      const int c = k0 - 16;
      ah[2*j]   = (_Float16)(H[rowA*16 + c    ] * ldsR[wave*256 + m*16 + c]);
      ah[2*j+1] = (_Float16)(H[rowA*16 + c + 1] * ldsR[wave*256 + m*16 + c + 1]);
    }
  }
  v8f cih;
  const float lhv = lbh[m];
#pragma unroll
  for (int j = 0; j < 8; ++j) cih[j] = lhv;

  v8f Hd = __builtin_amdgcn_wmma_f32_16x16x32_f16(false, ah, false, bhf,
                                                  (short)0, cih, false, false);

  if (valid) {
#pragma unroll
    for (int j = 0; j < 8; ++j) {
      int row = rowBase + j + hi * 8;
      float hold = H[row * 16 + m];
      float ht = fast_tanh(Hd[j]);
      Hn[row * 16 + m] = zs[j] * hold + (1.f - zs[j]) * ht;
    }
  }
}

// ---------------- head: (4 x 160000) @ (160000 x 512), K-split partials -----
__global__ void k_head(const float* __restrict__ H, const float* __restrict__ W1,
                       float* __restrict__ acc) {
  int j = blockIdx.x * 256 + threadIdx.x;    // 0..511 (grid.x == 2)
  int k0 = blockIdx.y * 1600;                // 100 K-chunks
  float a0 = 0.f, a1 = 0.f, a2 = 0.f, a3 = 0.f;
  for (int k = k0; k < k0 + 1600; ++k) {
    __builtin_prefetch(&W1[(size_t)(k + 16) * 512 + j], 0, 1);
    float w = W1[(size_t)k * 512 + j];
    a0 += H[k]            * w;
    a1 += H[160000 + k]   * w;
    a2 += H[320000 + k]   * w;
    a3 += H[480000 + k]   * w;
  }
  atomicAdd(&acc[j],        a0);
  atomicAdd(&acc[512 + j],  a1);
  atomicAdd(&acc[1024 + j], a2);
  atomicAdd(&acc[1536 + j], a3);
}

__global__ void k_final(const float* __restrict__ acc, const float* __restrict__ b1,
                        const float* __restrict__ W2, const float* __restrict__ b2,
                        float* __restrict__ out) {
  int tid = threadIdx.x;
  if (tid >= NB * 2) return;
  int b = tid >> 1, o = tid & 1;
  float s = b2[o];
  for (int j = 0; j < 512; ++j) {
    float h = acc[b * 512 + j] + b1[j];
    h = h > 0.f ? h : 0.01f * h;               // LeakyReLU(0.01)
    s += h * W2[j * 2 + o];
  }
  out[b * 2 + o] = s;
}

// ---------------- launcher ----------------
extern "C" void kernel_launch(void* const* d_in, const int* in_sizes, int n_in,
                              void* d_out, int out_size, void* d_ws, size_t ws_size,
                              hipStream_t stream) {
  const float*     x   = (const float*)d_in[0];
  const long long* ei  = (const long long*)d_in[1];   // (2, E) int64
  const float*     ew  = (const float*)d_in[2];
  const float *Wz = (const float*)d_in[3],  *bz  = (const float*)d_in[4];
  const float *Lz = (const float*)d_in[5],  *lbz = (const float*)d_in[6];
  const float *Wr = (const float*)d_in[7],  *br  = (const float*)d_in[8];
  const float *Lr = (const float*)d_in[9],  *lbr = (const float*)d_in[10];
  const float *Wh = (const float*)d_in[11], *bh  = (const float*)d_in[12];
  const float *Lh = (const float*)d_in[13], *lbh = (const float*)d_in[14];
  const float *W1 = (const float*)d_in[15], *b1  = (const float*)d_in[16];
  const float *W2 = (const float*)d_in[17], *b2  = (const float*)d_in[18];
  float* out = (float*)d_out;
  const int E = in_sizes[2];

  // workspace carve-out (256B aligned)
  char* wsb = (char*)d_ws;
  size_t off = 0;
  auto carve = [&](size_t bytes) -> char* {
    char* p = wsb + off;
    off = (off + bytes + 255) & ~(size_t)255;
    return p;
  };
  float*    deg  = (float*)carve((size_t)NN * 4);
  float*    nrm  = (float*)carve((size_t)(E + NN) * 4);
  _Float16* LT   = (_Float16*)carve(3 * 512 * 2);
  float*    sbuf = (float*)carve((size_t)3 * HSZ * 4);   // sz|sr|sh contiguous
  float*    cbuf = (float*)carve((size_t)3 * HSZ * 4);   // cz|cr|ch contiguous
  float*    Ha   = (float*)carve((size_t)HSZ * 4);
  float*    Hb   = (float*)carve((size_t)HSZ * 4);
  float*    acc  = (float*)carve((size_t)NB * 512 * 4);
  float *sz = sbuf, *sr = sbuf + HSZ, *sh = sbuf + 2 * HSZ;
  float *cz = cbuf, *cr = cbuf + HSZ, *ch = cbuf + 2 * HSZ;

  // graph normalization (self-loop weight 1 folded into deg init)
  k_fill<<<64, 256, 0, stream>>>(deg, 1.0f, NN);
  k_degscatter<<<(E + 255) / 256, 256, 0, stream>>>(ei, ew, deg, E);
  k_rsqrt<<<(NN + 255) / 256, 256, 0, stream>>>(deg);
  k_norm<<<(E + NN + 255) / 256, 256, 0, stream>>>(ei, ew, deg, nrm, E);
  k_cvtw<<<6, 256, 0, stream>>>(Lz, Lr, Lh, LT);

  float* Hcur = Ha;
  float* Hnxt = Hb;
  k_fill<<<1024, 256, 0, stream>>>(Hcur, 0.f, HSZ);

  const int totScatter = (E + NN) * CC;
  for (int t = 0; t < TT; ++t) {
    k_xw<<<(BN + 255) / 256, 256, 0, stream>>>(x, Wz, Wr, Wh, sz, sr, sh, t);
    k_fill<<<2048, 256, 0, stream>>>(cbuf, 0.f, 3 * HSZ);
    k_scatter<<<(totScatter + 255) / 256, 256, 0, stream>>>(
        ei, nrm, sz, sr, sh, cz, cr, ch, E);
    k_gru<<<(NTILE + 7) / 8, 256, 0, stream>>>(
        cz, cr, ch, bz, br, bh, LT, lbz, lbr, lbh, Hcur, Hnxt);
    float* tmp = Hcur; Hcur = Hnxt; Hnxt = tmp;
  }

  k_fill<<<8, 256, 0, stream>>>(acc, 0.f, NB * 512);
  dim3 hg(2, 100);
  k_head<<<hg, 256, 0, stream>>>(Hcur, W1, acc);
  k_final<<<1, 32, 0, stream>>>(acc, b1, W2, b2, out);
}
